// MoE_80410377716151
// MI455X (gfx1250) — matrix operand order repaired and checked by reading, not compiled, not run on hardware
//
#include <hip/hip_runtime.h>
#include <hip/hip_bf16.h>
#include <math.h>

typedef __attribute__((ext_vector_type(16))) __bf16 bf16x16;
typedef __attribute__((ext_vector_type(8)))  __bf16 bf16x8;
typedef __attribute__((ext_vector_type(4)))  __bf16 bf16x4;
typedef __attribute__((ext_vector_type(8)))  float  f32x8;
typedef __attribute__((ext_vector_type(4)))  float  f32x4;

#define DIMC    2048
#define INTERC  1024
#define NEXP    32
#define NTOK    4096
#define ASTR    40      /* LDS stride (bf16 elems) for 32-wide K tiles */
#define HSTR_R  1032    /* routed h buffer stride (1024 + pad) */
#define HSTR_S  2056    /* shared h buffer stride (2048 + pad) */

__device__ __forceinline__ f32x8 wmma_bf16(bf16x16 a, bf16x16 b, f32x8 c) {
  return __builtin_amdgcn_wmma_f32_16x16x32_bf16(
      /*neg_a=*/false, a, /*neg_b=*/false, b,
      /*c_mod=*/(short)0, c, /*reuse_a=*/false, /*reuse_b=*/false);
}

__device__ __forceinline__ f32x8 fzero8() {
  f32x8 z;
#pragma unroll
  for (int i = 0; i < 8; ++i) z[i] = 0.f;
  return z;
}

// Force operands into distinct live registers (defeats temp-block recycling
// that serializes ds_load -> wait -> wmma chains).
#define KEEP_LIVE4(a, b, c, d) \
  asm volatile("" : "+v"(a), "+v"(b), "+v"(c), "+v"(d))
#define KEEP_LIVE6(a, b, c, d, e, f) \
  asm volatile("" : "+v"(a), "+v"(b), "+v"(c), "+v"(d), "+v"(e), "+v"(f))

// A fragment (16x32 bf16, M x K): lane holds row m = lane&15; K chunks
// [half*8, half*8+8) and [16+half*8, 16+half*8+8)  (ISA 7.12.2)
__device__ __forceinline__ bf16x16 load_a_frag(const __bf16* base, int row0,
                                               int stride, int lane) {
  int m = lane & 15, half = lane >> 4;
  const __bf16* p = base + (size_t)(row0 + m) * stride + half * 8;
  bf16x8 lo = *(const bf16x8*)p;
  bf16x8 hi = *(const bf16x8*)(p + 16);
  bf16x16 r;
#pragma unroll
  for (int i = 0; i < 8; ++i) { r[i] = lo[i]; r[i + 8] = hi[i]; }
  return r;
}

// B fragment (32x16 bf16, K x N) staged as B^T (rows = N, K-contiguous):
// lane holds col n = lane&15, K chunks [half*8, half*8+8) and [16+half*8, 16+half*8+8)
__device__ __forceinline__ bf16x16 load_b_frag(const __bf16* base, int row0,
                                               int stride, int lane) {
  int nn = lane & 15, half = lane >> 4;
  const __bf16* p = base + (size_t)(row0 + nn) * stride + half * 8;
  bf16x8 lo = *(const bf16x8*)p;
  bf16x8 hi = *(const bf16x8*)(p + 16);
  bf16x16 r;
#pragma unroll
  for (int i = 0; i < 8; ++i) { r[i] = lo[i]; r[i + 8] = hi[i]; }
  return r;
}

// Register staging of a 128x32 f32 weight tile (two rows per thread).
struct WPref { f32x4 r0lo, r0hi, r1lo, r1hi; };

__device__ __forceinline__ void load_w_regs(const float* __restrict__ W,
                                            int ldw, int tid, WPref& p) {
  int cg = tid & 3, r0 = tid >> 2;
  const float* s0 = W + (size_t)r0 * ldw + cg * 8;
  const float* s1 = W + (size_t)(r0 + 64) * ldw + cg * 8;
  p.r0lo = *(const f32x4*)s0; p.r0hi = *(const f32x4*)(s0 + 4);
  p.r1lo = *(const f32x4*)s1; p.r1hi = *(const f32x4*)(s1 + 4);
}

__device__ __forceinline__ void store_w_regs(__bf16* lds, int tid,
                                             const WPref& p) {
  int cg = tid & 3, r0 = tid >> 2;
  bf16x8 o0, o1;
#pragma unroll
  for (int i = 0; i < 4; ++i) {
    o0[i] = (__bf16)p.r0lo[i]; o0[i + 4] = (__bf16)p.r0hi[i];
    o1[i] = (__bf16)p.r1lo[i]; o1[i + 4] = (__bf16)p.r1hi[i];
  }
  *(bf16x8*)(lds + r0 * ASTR + cg * 8) = o0;
  *(bf16x8*)(lds + (size_t)(r0 + 64) * ASTR + cg * 8) = o1;
}

// ---------------- Kernel 1: f32 -> bf16 convert (+ zero counters) ----------
__global__ __launch_bounds__(256) void moe_convert_kernel(
    const float* __restrict__ x, __bf16* __restrict__ xb,
    int* __restrict__ counts, size_t n) {
  if (blockIdx.x == 0 && threadIdx.x < NEXP) counts[threadIdx.x] = 0;
  size_t i = ((size_t)blockIdx.x * 256 + threadIdx.x) * 4;
  if (i < n) {
    f32x4 v = *(const f32x4*)(x + i);
    bf16x4 o;
#pragma unroll
    for (int j = 0; j < 4; ++j) o[j] = (__bf16)v[j];
    *(bf16x4*)(xb + i) = o;
  }
}

// ---------------- Kernel 2: gating (softmax + top-4 routing) ---------------
__global__ __launch_bounds__(256) void moe_gate_kernel(
    const float* __restrict__ x, const float* __restrict__ gw,
    int* __restrict__ counts, int* __restrict__ slot_tok,
    float* __restrict__ slot_w) {
  int wid = threadIdx.x >> 5, lane = threadIdx.x & 31;
  int t = blockIdx.x * 8 + wid;                  // one wave per token
  const f32x4* xr = (const f32x4*)(x + (size_t)t * DIMC);
  const f32x4* gr = (const f32x4*)(gw + (size_t)lane * DIMC);
  float acc = 0.f;
  for (int i = 0; i < DIMC / 4; ++i) {
    f32x4 a = xr[i], b = gr[i];
    acc += a[0] * b[0] + a[1] * b[1] + a[2] * b[2] + a[3] * b[3];
  }
  // softmax across 32 lanes (lane == expert)
  float m = acc;
  for (int off = 16; off; off >>= 1) m = fmaxf(m, __shfl_xor(m, off));
  float p = __expf(acc - m);
  float s = p;
  for (int off = 16; off; off >>= 1) s += __shfl_xor(s, off);
  float prob = p / s;
  // top-4 by iterated wave argmax
  float v = prob;
#pragma unroll
  for (int k = 0; k < 4; ++k) {
    float mv = v; int mi = lane;
    for (int off = 16; off; off >>= 1) {
      float ov = __shfl_xor(mv, off);
      int   oi = __shfl_xor(mi, off);
      if (ov > mv || (ov == mv && oi < mi)) { mv = ov; mi = oi; }
    }
    if (lane == mi) {
      int slot = atomicAdd(&counts[lane], 1);
      slot_tok[lane * NTOK + slot] = t;
      slot_w[lane * NTOK + slot] = prob * 1.0f;  // ROUTE_SCALE
      v = -INFINITY;
    }
  }
}

// ---------------- Kernel 3: shared SwiGLU expert (writes out = z) ----------
__global__ __launch_bounds__(256) void moe_shared_ffn_kernel(
    const __bf16* __restrict__ xb, const float* __restrict__ sw1,
    const float* __restrict__ sw2, const float* __restrict__ sw3,
    float* __restrict__ out) {
  __shared__ __align__(16) __bf16 hbuf[32 * HSTR_S];
  __shared__ __align__(16) __bf16 atile[2][32 * ASTR];
  __shared__ __align__(16) __bf16 btileA[2][128 * ASTR];
  __shared__ __align__(16) __bf16 btileB[2][128 * ASTR];
  int tid = threadIdx.x, wid = tid >> 5, lane = tid & 31;
  int wm = (wid & 1) * 16, wn = (wid >> 1) * 32;  // wave: 16x32 sub-tile
  int t0 = blockIdx.x * 32;
  int nn = lane & 15, half = lane >> 4;
  int ar = tid >> 2, acg = tid & 3;
  const __bf16* asrc = xb + (size_t)(t0 + ar) * DIMC + acg * 8;

  // phase 1: h[32, 2048] = silu(x @ sw1^T) * (x @ sw3^T), kept in LDS
  for (int it = 0; it < 16; ++it) {
    const float* W1 = sw1 + (size_t)(it * 128) * DIMC;
    const float* W3 = sw3 + (size_t)(it * 128) * DIMC;
    f32x8 acc1[2] = {fzero8(), fzero8()};
    f32x8 acc3[2] = {fzero8(), fzero8()};
    WPref p1, p3; bf16x8 pa;
    // prologue: fill buffer 0 with K-step 0
    if (tid < 128) pa = *(const bf16x8*)(asrc);
    load_w_regs(W1, DIMC, tid, p1);
    load_w_regs(W3, DIMC, tid, p3);
    if (tid < 128) *(bf16x8*)&atile[0][ar * ASTR + acg * 8] = pa;
    store_w_regs(btileA[0], tid, p1);
    store_w_regs(btileB[0], tid, p3);
    __syncthreads();
    for (int ks = 0; ks < DIMC / 32; ++ks) {
      int cur = ks & 1, nxt = cur ^ 1;
      int kn = (ks + 1 < DIMC / 32 ? ks + 1 : ks) * 32;
      // issue next K-step's global loads (no barrier crossed before use)
      if (tid < 128) pa = *(const bf16x8*)(asrc + kn);
      load_w_regs(W1 + kn, DIMC, tid, p1);
      load_w_regs(W3 + kn, DIMC, tid, p3);
      // preload ALL fragments into distinct regs, then back-to-back WMMAs
      bf16x16 a   = load_a_frag(atile[cur], wm, ASTR, lane);
      bf16x16 b10 = load_b_frag(btileA[cur], wn, ASTR, lane);
      bf16x16 b11 = load_b_frag(btileA[cur], wn + 16, ASTR, lane);
      bf16x16 b30 = load_b_frag(btileB[cur], wn, ASTR, lane);
      bf16x16 b31 = load_b_frag(btileB[cur], wn + 16, ASTR, lane);
      KEEP_LIVE4(b10, b11, b30, b31);
      acc1[0] = wmma_bf16(a, b10, acc1[0]);
      acc1[1] = wmma_bf16(a, b11, acc1[1]);
      acc3[0] = wmma_bf16(a, b30, acc3[0]);
      acc3[1] = wmma_bf16(a, b31, acc3[1]);
      // commit next K-step into buf[nxt]
      if (tid < 128) *(bf16x8*)&atile[nxt][ar * ASTR + acg * 8] = pa;
      store_w_regs(btileA[nxt], tid, p1);
      store_w_regs(btileB[nxt], tid, p3);
      __syncthreads();
    }
#pragma unroll
    for (int ni = 0; ni < 2; ++ni)
#pragma unroll
      for (int r = 0; r < 8; ++r) {
        int row = wm + half * 8 + r;
        int col = it * 128 + wn + ni * 16 + nn;
        float g = acc1[ni][r], u = acc3[ni][r];
        hbuf[row * HSTR_S + col] = (__bf16)(g / (1.f + __expf(-g)) * u);
      }
  }
  __syncthreads();
  // phase 2: z[32, 2048] = h @ sw2^T  (K = 2048, A from LDS h)
  for (int ot = 0; ot < 16; ++ot) {
    const float* W2 = sw2 + (size_t)(ot * 128) * (2 * INTERC);
    f32x8 acc[2] = {fzero8(), fzero8()};
    WPref p2;
    load_w_regs(W2, 2 * INTERC, tid, p2);
    store_w_regs(btileA[0], tid, p2);
    __syncthreads();
    for (int ks = 0; ks < (2 * INTERC) / 32; ++ks) {
      int cur = ks & 1, nxt = cur ^ 1;
      int kn = (ks + 1 < (2 * INTERC) / 32 ? ks + 1 : ks) * 32;
      load_w_regs(W2 + kn, 2 * INTERC, tid, p2);
      bf16x16 a  = load_a_frag(hbuf + ks * 32, wm, HSTR_S, lane);
      bf16x16 b0 = load_b_frag(btileA[cur], wn, ASTR, lane);
      bf16x16 b1 = load_b_frag(btileA[cur], wn + 16, ASTR, lane);
      asm volatile("" : "+v"(a), "+v"(b0), "+v"(b1));
      acc[0] = wmma_bf16(a, b0, acc[0]);
      acc[1] = wmma_bf16(a, b1, acc[1]);
      store_w_regs(btileA[nxt], tid, p2);
      __syncthreads();
    }
#pragma unroll
    for (int ni = 0; ni < 2; ++ni)
#pragma unroll
      for (int r = 0; r < 8; ++r) {
        int row = t0 + wm + half * 8 + r;
        int col = ot * 128 + wn + ni * 16 + nn;
        out[(size_t)row * DIMC + col] = acc[ni][r];
      }
  }
}

// ---------------- Kernel 4: routed experts (atomic-add into out) -----------
__global__ __launch_bounds__(256) void moe_routed_ffn_kernel(
    const __bf16* __restrict__ xb, const float* __restrict__ w1,
    const float* __restrict__ w2, const float* __restrict__ w3,
    const int* __restrict__ counts, const int* __restrict__ slot_tok,
    const float* __restrict__ slot_w, float* __restrict__ out) {
  __shared__ __align__(16) __bf16 hbuf[64 * HSTR_R];
  __shared__ __align__(16) __bf16 atile[2][64 * ASTR];
  __shared__ __align__(16) __bf16 btileA[2][128 * ASTR];
  __shared__ __align__(16) __bf16 btileB[2][128 * ASTR];
  __shared__ int   stok[64];
  __shared__ float swt[64];

  int e = blockIdx.x;
  int tile = blockIdx.y;
  int n = counts[e];
  if (tile * 64 >= n) return;
  int tid = threadIdx.x, wid = tid >> 5, lane = tid & 31;
  int wm = (wid & 1) * 32, wn = (wid >> 1) * 32;  // wave: 32x32 sub-tile
  int nn = lane & 15, half = lane >> 4;

  if (tid < 64) {
    int s = tile * 64 + tid;
    if (s < n) { stok[tid] = slot_tok[e * NTOK + s]; swt[tid] = slot_w[e * NTOK + s]; }
    else       { stok[tid] = 0;                      swt[tid] = 0.f; }
  }
  __syncthreads();
  const float* W1e = w1 + (size_t)e * INTERC * DIMC;
  const float* W3e = w3 + (size_t)e * INTERC * DIMC;
  const float* W2e = w2 + (size_t)e * DIMC * INTERC;
  int ar = tid >> 2, acg = tid & 3;
  const __bf16* asrc = xb + (size_t)stok[ar] * DIMC + acg * 8;  // gather row

  // phase 1: h[64, 1024]
  for (int it = 0; it < 8; ++it) {
    const float* W1 = W1e + (size_t)(it * 128) * DIMC;
    const float* W3 = W3e + (size_t)(it * 128) * DIMC;
    f32x8 acc1[2][2] = {{fzero8(), fzero8()}, {fzero8(), fzero8()}};
    f32x8 acc3[2][2] = {{fzero8(), fzero8()}, {fzero8(), fzero8()}};
    WPref p1, p3; bf16x8 pa;
    pa = *(const bf16x8*)(asrc);
    load_w_regs(W1, DIMC, tid, p1);
    load_w_regs(W3, DIMC, tid, p3);
    *(bf16x8*)&atile[0][ar * ASTR + acg * 8] = pa;
    store_w_regs(btileA[0], tid, p1);
    store_w_regs(btileB[0], tid, p3);
    __syncthreads();
    for (int ks = 0; ks < DIMC / 32; ++ks) {
      int cur = ks & 1, nxt = cur ^ 1;
      int kn = (ks + 1 < DIMC / 32 ? ks + 1 : ks) * 32;
      pa = *(const bf16x8*)(asrc + kn);
      load_w_regs(W1 + kn, DIMC, tid, p1);
      load_w_regs(W3 + kn, DIMC, tid, p3);
      // preload ALL fragments into distinct regs, then back-to-back WMMAs
      bf16x16 a0  = load_a_frag(atile[cur], wm, ASTR, lane);
      bf16x16 a1  = load_a_frag(atile[cur], wm + 16, ASTR, lane);
      bf16x16 b10 = load_b_frag(btileA[cur], wn, ASTR, lane);
      bf16x16 b11 = load_b_frag(btileA[cur], wn + 16, ASTR, lane);
      bf16x16 b30 = load_b_frag(btileB[cur], wn, ASTR, lane);
      bf16x16 b31 = load_b_frag(btileB[cur], wn + 16, ASTR, lane);
      KEEP_LIVE6(a0, a1, b10, b11, b30, b31);
      acc1[0][0] = wmma_bf16(a0, b10, acc1[0][0]);
      acc1[1][0] = wmma_bf16(a1, b10, acc1[1][0]);
      acc1[0][1] = wmma_bf16(a0, b11, acc1[0][1]);
      acc1[1][1] = wmma_bf16(a1, b11, acc1[1][1]);
      acc3[0][0] = wmma_bf16(a0, b30, acc3[0][0]);
      acc3[1][0] = wmma_bf16(a1, b30, acc3[1][0]);
      acc3[0][1] = wmma_bf16(a0, b31, acc3[0][1]);
      acc3[1][1] = wmma_bf16(a1, b31, acc3[1][1]);
      *(bf16x8*)&atile[nxt][ar * ASTR + acg * 8] = pa;
      store_w_regs(btileA[nxt], tid, p1);
      store_w_regs(btileB[nxt], tid, p3);
      __syncthreads();
    }
#pragma unroll
    for (int mi = 0; mi < 2; ++mi)
#pragma unroll
      for (int ni = 0; ni < 2; ++ni)
#pragma unroll
        for (int r = 0; r < 8; ++r) {
          int row = wm + mi * 16 + half * 8 + r;
          int col = it * 128 + wn + ni * 16 + nn;
          float g = acc1[mi][ni][r], u = acc3[mi][ni][r];
          hbuf[row * HSTR_R + col] = (__bf16)(g / (1.f + __expf(-g)) * u);
        }
  }
  __syncthreads();
  // phase 2: out += (h @ w2^T) * weight   (K = 1024, A from LDS h)
  for (int ot = 0; ot < 16; ++ot) {
    const float* W2 = W2e + (size_t)(ot * 128) * INTERC;
    f32x8 acc[2][2] = {{fzero8(), fzero8()}, {fzero8(), fzero8()}};
    WPref p2;
    load_w_regs(W2, INTERC, tid, p2);
    store_w_regs(btileA[0], tid, p2);
    __syncthreads();
    for (int ks = 0; ks < INTERC / 32; ++ks) {
      int cur = ks & 1, nxt = cur ^ 1;
      int kn = (ks + 1 < INTERC / 32 ? ks + 1 : ks) * 32;
      load_w_regs(W2 + kn, INTERC, tid, p2);
      bf16x16 a0 = load_a_frag(hbuf + ks * 32, wm, HSTR_R, lane);
      bf16x16 a1 = load_a_frag(hbuf + ks * 32, wm + 16, HSTR_R, lane);
      bf16x16 b0 = load_b_frag(btileA[cur], wn, ASTR, lane);
      bf16x16 b1 = load_b_frag(btileA[cur], wn + 16, ASTR, lane);
      KEEP_LIVE4(a0, a1, b0, b1);
      acc[0][0] = wmma_bf16(a0, b0, acc[0][0]);
      acc[1][0] = wmma_bf16(a1, b0, acc[1][0]);
      acc[0][1] = wmma_bf16(a0, b1, acc[0][1]);
      acc[1][1] = wmma_bf16(a1, b1, acc[1][1]);
      store_w_regs(btileA[nxt], tid, p2);
      __syncthreads();
    }
#pragma unroll
    for (int mi = 0; mi < 2; ++mi)
#pragma unroll
      for (int ni = 0; ni < 2; ++ni)
#pragma unroll
        for (int r = 0; r < 8; ++r) {
          int row = wm + mi * 16 + half * 8 + r;
          int col = ot * 128 + wn + ni * 16 + nn;
          float v = acc[mi][ni][r] * swt[row];
          unsafeAtomicAdd(&out[(size_t)stok[row] * DIMC + col], v);
        }
  }
}

extern "C" void kernel_launch(void* const* d_in, const int* in_sizes, int n_in,
                              void* d_out, int out_size, void* d_ws, size_t ws_size,
                              hipStream_t stream) {
  const float* x   = (const float*)d_in[0];
  const float* gw  = (const float*)d_in[1];
  const float* w1  = (const float*)d_in[2];
  const float* w2  = (const float*)d_in[3];
  const float* w3  = (const float*)d_in[4];
  const float* sw1 = (const float*)d_in[5];
  const float* sw2 = (const float*)d_in[6];
  const float* sw3 = (const float*)d_in[7];
  float* out = (float*)d_out;

  char* ws = (char*)d_ws;
  size_t xb_bytes = (size_t)NTOK * DIMC * sizeof(__bf16);   // 16 MiB
  __bf16* xb      = (__bf16*)ws;
  int*    counts  = (int*)(ws + xb_bytes);                  // 32 ints
  int*    stok    = (int*)(ws + xb_bytes + 256);            // 32*4096 ints
  float*  swt     = (float*)(ws + xb_bytes + 256 + (size_t)NEXP * NTOK * 4);

  size_t nelem = (size_t)NTOK * DIMC;
  moe_convert_kernel<<<dim3((unsigned)((nelem / 4 + 255) / 256)), 256, 0, stream>>>(
      x, xb, counts, nelem);
  moe_gate_kernel<<<dim3(NTOK / 8), 256, 0, stream>>>(x, gw, counts, stok, swt);
  moe_shared_ffn_kernel<<<dim3(NTOK / 32), 256, 0, stream>>>(xb, sw1, sw2, sw3, out);
  moe_routed_ffn_kernel<<<dim3(NEXP, NTOK / 64), 256, 0, stream>>>(
      xb, w1, w2, w3, counts, stok, swt, out);
}
